// TreeTransformer_79233556677041
// MI455X (gfx1250) — matrix operand order repaired and checked by reading, not compiled, not run on hardware
//
#include <hip/hip_runtime.h>
#include <hip/hip_bf16.h>

// ---------------------------------------------------------------------------
// TreeTransformer on MI455X (gfx1250, wave32, WMMA).
// All GEMMs: bf16 inputs, v_wmma_f32_16x16x32_bf16, f32 accumulate.
// A-tiles staged to LDS with GLOBAL_LOAD_ASYNC_TO_LDS_B128 (ASYNCcnt path).
// Attention: flash-style online softmax, mask from dist on the fly.
// ---------------------------------------------------------------------------

typedef __attribute__((ext_vector_type(16))) __bf16 v16bf;
typedef __attribute__((ext_vector_type(8)))  __bf16 v8bf;
typedef __attribute__((ext_vector_type(8)))  float  v8f;

#define NPOS 1024
#define BATCH 8
#define NTOK  8192      // NPOS*BATCH, row index = n*B + b
#define HID   256
#define NH    8
#define DH    32
#define EPSF  1e-5f

#define FLAG_RELU 1
#define FLAG_LN   2

__device__ __forceinline__ unsigned short f32_to_bf16(float x) {
  unsigned int u = __float_as_uint(x);
  unsigned int r = (u + 0x7FFFu + ((u >> 16) & 1u)) >> 16;
  return (unsigned short)r;
}

// ---- WMMA fragment loaders (contiguous k order, same in A and B) ------

// A fragment from an LDS-resident strip s[16][ldk] at column k0.
// lane: m = lane&15, kh = lane>>4. halves: k = {kh*16..+7, kh*16+8..+15}
__device__ __forceinline__ v16bf load_a_frag_lds(const unsigned short* __restrict__ s,
                                                 int ldk, int k0, int lane) {
  int m = lane & 15, kh = lane >> 4;
  const unsigned short* p = s + m * ldk + k0 + kh * 16;
  union { v16bf v; v8bf h[2]; } r;
  r.h[0] = *(const v8bf*)(p);
  r.h[1] = *(const v8bf*)(p + 8);
  return r.v;
}

// B fragment: 32x16 bf16 from pre-TRANSPOSED weights wt[Ncols][K].
// lane: n = lane&15 (col), kh = lane>>4; k = kh*16 + 2j + t  -> contiguous.
__device__ __forceinline__ v16bf load_bT_frag(const unsigned short* __restrict__ wt,
                                              int ldk, int col0, int k0, int lane) {
  int n = lane & 15, kh = lane >> 4;
  return *(const v16bf*)(wt + (size_t)(col0 + n) * ldk + k0 + kh * 16);
}

// ---------------------------------------------------------------------------
// prep: t = nodes + sinusoidal PE, store bf16  (rows = n*B+b, cols = 256)
// ---------------------------------------------------------------------------
__global__ void prep_embed_kernel(const float* __restrict__ nodes,
                                  unsigned short* __restrict__ tbf) {
  int row = blockIdx.x;          // 0..8191
  int c   = threadIdx.x;         // 0..255
  int n   = row >> 3;            // position
  float i2    = (float)((c >> 1) << 1);
  float expo  = i2 * (1.0f / 256.0f);
  float denom = __expf(expo * 9.210340371976184f);  // 10000^expo
  float ang   = (float)n / denom;
  float pe    = (c & 1) ? __cosf(ang) : __sinf(ang);
  float v = nodes[(size_t)row * HID + c] + pe;
  tbf[(size_t)row * HID + c] = f32_to_bf16(v);
}

// f32 [K][Ncols] -> bf16 transposed [Ncols][K]
__global__ void cvt_T_kernel(const float* __restrict__ src,
                             unsigned short* __restrict__ dst, int K, int Ncols) {
  int i = blockIdx.x * 256 + threadIdx.x;
  if (i >= K * Ncols) return;
  int k = i / Ncols, c = i - k * Ncols;
  dst[(size_t)c * K + k] = f32_to_bf16(src[i]);
}

// ---------------------------------------------------------------------------
// Generic GEMM: out[M=8192][Ntot] = A_bf16[M][K] x W^T_bf16[Ntot][K] + bias
// block = 256 thr (8 waves); tile = 16 rows x 256 cols; wave w -> cols w*32..
// A strip (16 x K) staged once per block via async copy to LDS.
// Optional: +resid (ld 256), ReLU, LayerNorm over 256 cols (Ntot==256 only).
// ---------------------------------------------------------------------------
__global__ __launch_bounds__(256) void gemm_kernel(
    const unsigned short* __restrict__ A, const unsigned short* __restrict__ WT,
    const float* __restrict__ bias, const float* __restrict__ resid,
    const float* __restrict__ ln_g, const float* __restrict__ ln_b,
    float* __restrict__ out_f32, unsigned short* __restrict__ out_bf,
    int K, int Ntot, int flags) {
  __shared__ __align__(32) unsigned short aStrip[16 * 512];  // up to K=512
  __shared__ __align__(32) float tile[16 * 256];
  __shared__ float mu_s[16], rs_s[16];

  int lane = threadIdx.x & 31;
  int wave = threadIdx.x >> 5;
  int row0 = blockIdx.x * 16;
  int colblk = blockIdx.y * 256;
  int col0 = colblk + wave * 32;

  // ---- async-stage A strip: 16 rows x K bf16 -> LDS (b128 granules) ----
  {
    int total = (16 * K) >> 3;               // # of 16-byte transfers
    for (int i = threadIdx.x; i < total; i += 256) {
      int e = i << 3;                        // element offset in strip
      int row = e / K;
      int col = e - row * K;
      unsigned ldsoff = (unsigned)(size_t)(&aStrip[e]);   // low 32 bits = LDS addr
      const unsigned short* g = A + (size_t)(row0 + row) * K + col;
      asm volatile("global_load_async_to_lds_b128 %0, %1, off"
                   :: "v"(ldsoff), "v"(g) : "memory");
    }
    asm volatile("s_wait_asynccnt 0x0" ::: "memory");
    __syncthreads();
  }

  // ---- K loop, software-pipelined (next frags in flight during WMMA) ----
  v8f c0 = {}, c1 = {};
  v16bf a  = load_a_frag_lds(aStrip, K, 0, lane);
  v16bf b0 = load_bT_frag(WT, K, col0,      0, lane);
  v16bf b1 = load_bT_frag(WT, K, col0 + 16, 0, lane);
  for (int k0 = 0; k0 < K; k0 += 32) {
    int k1 = k0 + 32;
    v16bf a_n = a, b0_n = b0, b1_n = b1;
    if (k1 < K) {
      __builtin_prefetch(WT + (size_t)(col0 + (lane & 15)) * K + k1 + 32, 0, 3);
      a_n  = load_a_frag_lds(aStrip, K, k1, lane);
      b0_n = load_bT_frag(WT, K, col0,      k1, lane);
      b1_n = load_bT_frag(WT, K, col0 + 16, k1, lane);
    }
    c0 = __builtin_amdgcn_wmma_f32_16x16x32_bf16(false, a, false, b0, (short)0, c0, false, false);
    c1 = __builtin_amdgcn_wmma_f32_16x16x32_bf16(false, a, false, b1, (short)0, c1, false, false);
    a = a_n; b0 = b0_n; b1 = b1_n;
  }

  int n  = lane & 15;
  int mh = (lane >> 4) * 8;
  bool do_relu = (flags & FLAG_RELU) != 0;
  bool do_ln   = (flags & FLAG_LN) != 0;

  float vals[16];
#pragma unroll
  for (int r = 0; r < 8; ++r) {
    int m = mh + r;
    int g0 = col0 + n, g1 = col0 + 16 + n;
    float v0 = c0[r] + bias[g0];
    float v1 = c1[r] + bias[g1];
    if (resid) {
      v0 += resid[(size_t)(row0 + m) * Ntot + g0];
      v1 += resid[(size_t)(row0 + m) * Ntot + g1];
    }
    if (do_relu) { v0 = fmaxf(v0, 0.0f); v1 = fmaxf(v1, 0.0f); }
    vals[r] = v0; vals[8 + r] = v1;
  }

  if (do_ln) {
    // Ntot == 256, colblk == 0: block owns whole rows.
#pragma unroll
    for (int r = 0; r < 8; ++r) {
      int m = mh + r;
      tile[m * 256 + (col0 - colblk) + n]      = vals[r];
      tile[m * 256 + (col0 - colblk) + 16 + n] = vals[8 + r];
    }
    __syncthreads();
    if (threadIdx.x < 16) {
      float s = 0.0f, s2 = 0.0f;
      for (int c = 0; c < 256; ++c) {
        float x = tile[threadIdx.x * 256 + c];
        s += x; s2 += x * x;
      }
      float mu  = s * (1.0f / 256.0f);
      float var = s2 * (1.0f / 256.0f) - mu * mu;
      mu_s[threadIdx.x] = mu;
      rs_s[threadIdx.x] = rsqrtf(var + EPSF);
    }
    __syncthreads();
#pragma unroll
    for (int r = 0; r < 8; ++r) {
      int m = mh + r;
      int g0 = col0 + n, g1 = col0 + 16 + n;
      float o0 = (vals[r]     - mu_s[m]) * rs_s[m] * ln_g[g0] + ln_b[g0];
      float o1 = (vals[8 + r] - mu_s[m]) * rs_s[m] * ln_g[g1] + ln_b[g1];
      size_t i0 = (size_t)(row0 + m) * Ntot + g0;
      size_t i1 = (size_t)(row0 + m) * Ntot + g1;
      out_f32[i0] = o0; out_f32[i1] = o1;
      out_bf[i0] = f32_to_bf16(o0); out_bf[i1] = f32_to_bf16(o1);
    }
  } else {
#pragma unroll
    for (int r = 0; r < 8; ++r) {
      int m = mh + r;
      size_t i0 = (size_t)(row0 + m) * Ntot + col0 + n;
      size_t i1 = (size_t)(row0 + m) * Ntot + col0 + 16 + n;
      out_bf[i0] = f32_to_bf16(vals[r]);
      out_bf[i1] = f32_to_bf16(vals[8 + r]);
      if (out_f32) { out_f32[i0] = vals[r]; out_f32[i1] = vals[8 + r]; }
    }
  }
}

// ---------------------------------------------------------------------------
// Flash attention: one wave per (b, h, 16-query tile).
// qkv layout: rows n*B+b, 768 cols = [q(256) | k(256) | v(256)], bf16.
// ---------------------------------------------------------------------------
__device__ __forceinline__ v16bf load_kt_frag(const unsigned short* __restrict__ qkv,
                                              int b, int h, int key0, int lane) {
  int n = lane & 15, kh = lane >> 4;
  const unsigned short* p =
      qkv + (size_t)((key0 + n) * BATCH + b) * 768 + 256 + h * 32 + kh * 16;
  return *(const v16bf*)p;   // B[k=dh][col=key]: contiguous along dh
}

__device__ __forceinline__ v16bf load_v_frag(const unsigned short* __restrict__ qkv,
                                             int b, int h, int key0, int dhh, int lane) {
  int n = lane & 15, kh = lane >> 4;
  const unsigned short* base =
      qkv + (size_t)((key0 + kh * 16) * BATCH + b) * 768 + 512 + h * 32 + dhh * 16 + n;
  union { v16bf v; unsigned short u[16]; } r;
#pragma unroll
  for (int j = 0; j < 8; ++j) {
    r.u[2 * j]     = base[(size_t)(2 * j)     * (BATCH * 768)];
    r.u[2 * j + 1] = base[(size_t)(2 * j + 1) * (BATCH * 768)];
  }
  return r.v;
}

__global__ __launch_bounds__(32) void attn_kernel(
    const unsigned short* __restrict__ qkv, const int* __restrict__ dist,
    unsigned short* __restrict__ obf) {
  __shared__ __align__(32) unsigned short pshare[16 * 32];
  int lane = threadIdx.x;
  int qt = blockIdx.x;   // 0..63 query tile
  int h  = blockIdx.y;   // 0..7
  int b  = blockIdx.z;   // 0..7
  const float scale = 0.17677669529663687f;  // 1/sqrt(32)

  int mA = lane & 15, khA = lane >> 4;        // A-frag mapping
  int nC = lane & 15, mhC = (lane >> 4) * 8;  // C-frag mapping

  // Q fragment (16 queries x 32 dh)
  union { v16bf v; v8bf hv[2]; } qf;
  {
    const unsigned short* p =
        qkv + (size_t)((qt * 16 + mA) * BATCH + b) * 768 + h * 32 + khA * 16;
    qf.hv[0] = *(const v8bf*)(p);
    qf.hv[1] = *(const v8bf*)(p + 8);
  }

  float mrow[8], lrow[8];
  v8f o0 = {}, o1 = {};
#pragma unroll
  for (int r = 0; r < 8; ++r) { mrow[r] = -1e30f; lrow[r] = 0.0f; }

  for (int ct = 0; ct < 32; ++ct) {        // 32 key-pairs of 32 -> 1024 keys
    int key0 = ct * 32;
    v16bf bk0 = load_kt_frag(qkv, b, h, key0,      lane);
    v16bf bk1 = load_kt_frag(qkv, b, h, key0 + 16, lane);
    v8f z = {};
    v8f s0 = __builtin_amdgcn_wmma_f32_16x16x32_bf16(false, qf.v, false, bk0, (short)0, z, false, false);
    v8f s1 = __builtin_amdgcn_wmma_f32_16x16x32_bf16(false, qf.v, false, bk1, (short)0, z, false, false);

    float sv[16];
#pragma unroll
    for (int r = 0; r < 8; ++r) {
      int qrow = qt * 16 + mhC + r;
      int d0 = dist[((size_t)b * NPOS + qrow) * NPOS + key0 + nC];
      int d1 = dist[((size_t)b * NPOS + qrow) * NPOS + key0 + 16 + nC];
      bool a0 = (h >= 6) || ((h >= 2) ? (d0 >= 1) : (d0 == 1));
      bool a1 = (h >= 6) || ((h >= 2) ? (d1 >= 1) : (d1 == 1));
      sv[r]     = a0 ? s0[r] * scale : -1e30f;
      sv[8 + r] = a1 ? s1[r] * scale : -1e30f;
    }

    // online softmax per query row (row = 16 lanes of same half-wave)
#pragma unroll
    for (int r = 0; r < 8; ++r) {
      float mx = fmaxf(sv[r], sv[8 + r]);
      mx = fmaxf(mx, __shfl_xor(mx, 8, 32));
      mx = fmaxf(mx, __shfl_xor(mx, 4, 32));
      mx = fmaxf(mx, __shfl_xor(mx, 2, 32));
      mx = fmaxf(mx, __shfl_xor(mx, 1, 32));
      float mnew = fmaxf(mrow[r], mx);
      float f  = __expf(mrow[r] - mnew);
      float p0 = __expf(sv[r]     - mnew);
      float p1 = __expf(sv[8 + r] - mnew);
      float ps = p0 + p1;
      ps += __shfl_xor(ps, 8, 32);
      ps += __shfl_xor(ps, 4, 32);
      ps += __shfl_xor(ps, 2, 32);
      ps += __shfl_xor(ps, 1, 32);
      lrow[r] = lrow[r] * f + ps;
      mrow[r] = mnew;
      o0[r] *= f; o1[r] *= f;
      pshare[(mhC + r) * 32 + nC]      = f32_to_bf16(p0);
      pshare[(mhC + r) * 32 + 16 + nC] = f32_to_bf16(p1);
    }
    __syncthreads();

    // reload P with A-fragment layout (16 queries x 32 keys)
    union { v16bf v; v8bf hv[2]; } pf;
    {
      const unsigned short* pp = pshare + mA * 32 + khA * 16;
      pf.hv[0] = *(const v8bf*)(pp);
      pf.hv[1] = *(const v8bf*)(pp + 8);
    }
    __syncthreads();

    v16bf bv0 = load_v_frag(qkv, b, h, key0, 0, lane);
    v16bf bv1 = load_v_frag(qkv, b, h, key0, 1, lane);
    o0 = __builtin_amdgcn_wmma_f32_16x16x32_bf16(false, pf.v, false, bv0, (short)0, o0, false, false);
    o1 = __builtin_amdgcn_wmma_f32_16x16x32_bf16(false, pf.v, false, bv1, (short)0, o1, false, false);
  }

#pragma unroll
  for (int r = 0; r < 8; ++r) {
    int qrow = qt * 16 + mhC + r;
    float inv = 1.0f / lrow[r];
    size_t base = (size_t)(qrow * BATCH + b) * HID + h * 32;
    obf[base + nC]      = f32_to_bf16(o0[r] * inv);
    obf[base + 16 + nC] = f32_to_bf16(o1[r] * inv);
  }
}

// ---------------------------------------------------------------------------
// Final: batch-norm over B=8 rows of x[0]  (rows 0..7 of xf32)
// ---------------------------------------------------------------------------
__global__ void bn_kernel(const float* __restrict__ xf, const float* __restrict__ g,
                          const float* __restrict__ bb, float* __restrict__ out) {
  int c = threadIdx.x;   // 0..255
  float y[8];
  float s = 0.0f;
#pragma unroll
  for (int i = 0; i < 8; ++i) { y[i] = xf[i * HID + c]; s += y[i]; }
  float mu = s * 0.125f;
  float s2 = 0.0f;
#pragma unroll
  for (int i = 0; i < 8; ++i) { float d = y[i] - mu; s2 += d * d; }
  float rs = rsqrtf(s2 * 0.125f + EPSF);
#pragma unroll
  for (int i = 0; i < 8; ++i) out[i * HID + c] = (y[i] - mu) * rs * g[c] + bb[c];
}

// ---------------------------------------------------------------------------
extern "C" void kernel_launch(void* const* d_in, const int* in_sizes, int n_in,
                              void* d_out, int out_size, void* d_ws, size_t ws_size,
                              hipStream_t stream) {
  (void)in_sizes; (void)n_in; (void)out_size; (void)ws_size;
  const float* nodes  = (const float*)d_in[0];
  const int*   dist   = (const int*)  d_in[1];
  const float* W_in   = (const float*)d_in[2];
  const float* b_in   = (const float*)d_in[3];
  const float* ln_in_g= (const float*)d_in[4];
  const float* ln_in_b= (const float*)d_in[5];
  const float* Wqkv   = (const float*)d_in[6];
  const float* bqkv   = (const float*)d_in[7];
  const float* Wo     = (const float*)d_in[8];
  const float* bo     = (const float*)d_in[9];
  const float* ln1_g  = (const float*)d_in[10];
  const float* ln1_b  = (const float*)d_in[11];
  const float* W1     = (const float*)d_in[12];
  const float* b1     = (const float*)d_in[13];
  const float* W2     = (const float*)d_in[14];
  const float* b2     = (const float*)d_in[15];
  const float* ln2_g  = (const float*)d_in[16];
  const float* ln2_b  = (const float*)d_in[17];
  const float* bn_g   = (const float*)d_in[18];
  const float* bn_b   = (const float*)d_in[19];

  char* ws = (char*)d_ws;
  size_t off = 0;
  auto alloc = [&](size_t bytes) {
    off = (off + 255) & ~(size_t)255;
    void* p = ws + off;
    off += bytes;
    return p;
  };
  unsigned short* tbf    = (unsigned short*)alloc((size_t)NTOK * HID * 2);
  float*          xf32   = (float*)         alloc((size_t)NTOK * HID * 4);
  unsigned short* xbf    = (unsigned short*)alloc((size_t)NTOK * HID * 2);
  unsigned short* qkvbf  = (unsigned short*)alloc((size_t)NTOK * 768 * 2);
  unsigned short* obf    = (unsigned short*)alloc((size_t)NTOK * HID * 2);
  unsigned short* hbf    = (unsigned short*)alloc((size_t)NTOK * 512 * 2);
  unsigned short* wtin   = (unsigned short*)alloc((size_t)256 * 256 * 2);
  unsigned short* wtqkv  = (unsigned short*)alloc((size_t)2 * 768 * 256 * 2);
  unsigned short* wto    = (unsigned short*)alloc((size_t)2 * 256 * 256 * 2);
  unsigned short* wt1    = (unsigned short*)alloc((size_t)2 * 512 * 256 * 2);
  unsigned short* wt2    = (unsigned short*)alloc((size_t)2 * 256 * 512 * 2);

  // --- weight conversion (f32 -> bf16, transposed to [Ncols][K]) ---
  cvt_T_kernel<<<(256 * 256 + 255) / 256, 256, 0, stream>>>(W_in, wtin, 256, 256);
  for (int l = 0; l < 2; ++l) {
    cvt_T_kernel<<<(256 * 768 + 255) / 256, 256, 0, stream>>>(
        Wqkv + (size_t)l * 256 * 768, wtqkv + (size_t)l * 768 * 256, 256, 768);
    cvt_T_kernel<<<(256 * 256 + 255) / 256, 256, 0, stream>>>(
        Wo + (size_t)l * 256 * 256, wto + (size_t)l * 256 * 256, 256, 256);
    cvt_T_kernel<<<(256 * 512 + 255) / 256, 256, 0, stream>>>(
        W1 + (size_t)l * 256 * 512, wt1 + (size_t)l * 512 * 256, 256, 512);
    cvt_T_kernel<<<(512 * 256 + 255) / 256, 256, 0, stream>>>(
        W2 + (size_t)l * 512 * 256, wt2 + (size_t)l * 256 * 512, 512, 256);
  }

  // --- embedding: (nodes + PE) @ W_in + b_in, LayerNorm ---
  prep_embed_kernel<<<NTOK, 256, 0, stream>>>(nodes, tbf);
  gemm_kernel<<<dim3(NTOK / 16, 1), 256, 0, stream>>>(
      tbf, wtin, b_in, nullptr, ln_in_g, ln_in_b, xf32, xbf, 256, 256, FLAG_LN);

  for (int l = 0; l < 2; ++l) {
    // QKV projection (256 -> 768), bf16 out only
    gemm_kernel<<<dim3(NTOK / 16, 3), 256, 0, stream>>>(
        xbf, wtqkv + (size_t)l * 768 * 256, bqkv + (size_t)l * 768,
        nullptr, nullptr, nullptr, nullptr, qkvbf, 256, 768, 0);
    // masked multi-head attention
    attn_kernel<<<dim3(NPOS / 16, NH, BATCH), 32, 0, stream>>>(qkvbf, dist, obf);
    // output projection + residual + LN1
    gemm_kernel<<<dim3(NTOK / 16, 1), 256, 0, stream>>>(
        obf, wto + (size_t)l * 256 * 256, bo + (size_t)l * 256,
        xf32, ln1_g + (size_t)l * 256, ln1_b + (size_t)l * 256,
        xf32, xbf, 256, 256, FLAG_LN);
    // FFN1 (256 -> 512) + ReLU
    gemm_kernel<<<dim3(NTOK / 16, 2), 256, 0, stream>>>(
        xbf, wt1 + (size_t)l * 512 * 256, b1 + (size_t)l * 512,
        nullptr, nullptr, nullptr, nullptr, hbf, 256, 512, FLAG_RELU);
    // FFN2 (512 -> 256) + residual + LN2
    gemm_kernel<<<dim3(NTOK / 16, 1), 256, 0, stream>>>(
        hbf, wt2 + (size_t)l * 256 * 512, b2 + (size_t)l * 256,
        xf32, ln2_g + (size_t)l * 256, ln2_b + (size_t)l * 256,
        xf32, xbf, 512, 256, FLAG_LN);
  }

  // final batch-norm of position 0 -> d_out (8 x 256 f32)
  bn_kernel<<<1, 256, 0, stream>>>(xf32, bn_g, bn_b, (float*)d_out);
}
